// NoisyTopKRouter_2027224564195
// MI455X (gfx1250) — matrix operand order repaired and checked by reading, not compiled, not run on hardware
//
#include <hip/hip_runtime.h>
#include <math.h>

#define DIN   2048
#define NTOK  16384
#define NEXP  64
#define KTOP  8

#define BM 256         // block M tile
#define BN 128         // block N tile
#define BK 64          // K-tile; 2 WMMA K-steps of 32 per tile

typedef __attribute__((ext_vector_type(16))) __bf16 v16bf;
typedef __attribute__((ext_vector_type(8)))  float  v8f;
typedef __attribute__((ext_vector_type(8)))  float  f32x8;
typedef __attribute__((ext_vector_type(8)))  __bf16 bf16x8;

union BFrag16 {
    v16bf v;
    uint4 q[2];
};

__device__ __forceinline__ float gelu_exact(float v) {
    return 0.5f * v * (1.0f + erff(v * 0.70710678118654752f));
}

// CDNA5 async global->LDS copy (ASYNCcnt path, 16 bytes per lane)
__device__ __forceinline__ void async_ld_b128(unsigned lds_off, const void* gptr) {
    asm volatile("global_load_async_to_lds_b128 %0, %1, off"
                 :: "v"(lds_off), "v"((unsigned long long)(size_t)gptr)
                 : "memory");
}
__device__ __forceinline__ void wait_async_12() {
    asm volatile("s_wait_asynccnt 0xc" ::: "memory");
}
__device__ __forceinline__ void wait_async_0() {
    asm volatile("s_wait_asynccnt 0x0" ::: "memory");
}

// ---------------------------------------------------------------------------
// Conversion kernels (bandwidth-bound, one-shot): keep all fp32->bf16 VALU
// out of the GEMM hot loop.
// ---------------------------------------------------------------------------
__global__ __launch_bounds__(256) void cvt_x_kernel(const float* __restrict__ x,
                                                    unsigned short* __restrict__ xb) {
    size_t i = ((size_t)blockIdx.x * 256 + threadIdx.x) * 8;
    float4 a = *(const float4*)(x + i);
    float4 b = *(const float4*)(x + i + 4);
    f32x8 f = {a.x, a.y, a.z, a.w, b.x, b.y, b.z, b.w};
    union { bf16x8 v; uint4 q; } r;
    r.v = __builtin_convertvector(f, bf16x8);       // v_cvt_pk_bf16_f32
    *(uint4*)(xb + i) = r.q;
}

// W1 [k][n] fp32 -> w1t [n][k] bf16 (transposed so GEMM B-tiles are contiguous)
__global__ __launch_bounds__(256) void cvt_w1t_kernel(const float* __restrict__ W1,
                                                      unsigned short* __restrict__ w1t) {
    __shared__ float t[32][33];
    const int tx = threadIdx.x & 31;
    const int ty = threadIdx.x >> 5;            // 0..7
#pragma unroll
    for (int i = 0; i < 4; ++i) {
        int k = blockIdx.y * 32 + ty + i * 8;
        t[ty + i * 8][tx] = W1[(size_t)k * DIN + blockIdx.x * 32 + tx];
    }
    __syncthreads();
#pragma unroll
    for (int i = 0; i < 4; ++i) {
        int n = blockIdx.x * 32 + ty + i * 8;
        union { __bf16 b; unsigned short u; } c;
        c.b = (__bf16)t[tx][ty + i * 8];
        w1t[(size_t)n * DIN + blockIdx.y * 32 + tx] = c.u;
    }
}

// W2 [d][e] fp32 -> w2t [e][d] bf16
__global__ __launch_bounds__(256) void cvt_w2t_kernel(const float* __restrict__ W2,
                                                      unsigned short* __restrict__ w2t) {
    int i = blockIdx.x * 256 + threadIdx.x;     // 64*2048
    int e = i >> 11;
    int d = i & 2047;
    union { __bf16 b; unsigned short u; } c;
    c.b = (__bf16)W2[(size_t)d * NEXP + e];
    w2t[(size_t)e * DIN + d] = c.u;
}

// ---------------------------------------------------------------------------
// Kernel 1: H = gelu(xb @ w1t^T + b1), all-bf16 operands, WMMA fp32 accum.
// Block: 256x128 tile, 8 waves (4Mx2N), wave = 64x64 (16 accum tiles,
// ds:wmma issue ratio 1:1). Double-buffered LDS filled by
// GLOBAL_LOAD_ASYNC_TO_LDS_B128 with one K-tile of lookahead.
// ---------------------------------------------------------------------------
__global__ __launch_bounds__(256) void gemm1_gelu(const unsigned short* __restrict__ xb,
                                                  const unsigned short* __restrict__ w1t,
                                                  const float* __restrict__ b1,
                                                  unsigned short* __restrict__ H) {
    __shared__ __align__(16) unsigned short As[2][BM * BK];   // 32 KB per buffer
    __shared__ __align__(16) unsigned short Bs[2][BN * BK];   // 16 KB per buffer

    const int tid  = threadIdx.x;
    const int lane = tid & 31;
    const int w    = tid >> 5;
    const int m0   = blockIdx.x * BM;
    const int n0   = blockIdx.y * BN;
    const int Moff = (w >> 1) * 64;            // wave grid 4(M) x 2(N)
    const int Noff = (w & 1) * 64;

    const int ml    = lane & 15;
    const int half  = lane >> 4;
    const int abase = half * 8;                // A-fragment K sub-base (ISA layout)
    const int bbase = half * 8;                // B-fragment K sub-base (ISA layout)

    // async-load assignment:
    //   A: 1 thread per row, 128 B (8 x b128)
    //   B: 2 threads per row, 64 B (4 x b128)
    const int brow  = tid >> 1;                // 0..127
    const int bkoff = (tid & 1) * 32;          // element offset 0 / 32

    const unsigned aldsb[2] = {(unsigned)(size_t)&As[0][tid * BK],
                               (unsigned)(size_t)&As[1][tid * BK]};
    const unsigned bldsb[2] = {(unsigned)(size_t)&Bs[0][brow * BK + bkoff],
                               (unsigned)(size_t)&Bs[1][brow * BK + bkoff]};
    const unsigned short* agp = xb  + (size_t)(m0 + tid) * DIN;
    const unsigned short* bgp = w1t + (size_t)(n0 + brow) * DIN + bkoff;

    v8f c[4][4] = {};
    const int ntiles = DIN / BK;               // 32

    // prologue: issue tile 0 into buffer 0 (12 async b128 per thread)
#pragma unroll
    for (int q = 0; q < 8; ++q)
        async_ld_b128(aldsb[0] + q * 16, (const char*)agp + q * 16);
#pragma unroll
    for (int q = 0; q < 4; ++q)
        async_ld_b128(bldsb[0] + q * 16, (const char*)bgp + q * 16);

    for (int kt = 0; kt < ntiles; ++kt) {
        const int cur = kt & 1;
        if (kt + 1 < ntiles) {
            const size_t gofs = (size_t)(kt + 1) * BK * 2;   // bytes along K
#pragma unroll
            for (int q = 0; q < 8; ++q)
                async_ld_b128(aldsb[cur ^ 1] + q * 16, (const char*)agp + gofs + q * 16);
#pragma unroll
            for (int q = 0; q < 4; ++q)
                async_ld_b128(bldsb[cur ^ 1] + q * 16, (const char*)bgp + gofs + q * 16);
            wait_async_12();                   // tile kt complete (in-order)
        } else {
            wait_async_0();
        }
        __syncthreads();

        const unsigned short* as = As[cur];
        const unsigned short* bs = Bs[cur];
#pragma unroll
        for (int ks = 0; ks < BK; ks += 32) {
            BFrag16 af[4];
#pragma unroll
            for (int i = 0; i < 4; ++i) {
                const int r = Moff + i * 16 + ml;
                af[i].q[0] = *(const uint4*)&as[r * BK + ks + abase];
                af[i].q[1] = *(const uint4*)&as[r * BK + ks + 16 + abase];
            }
#pragma unroll
            for (int j = 0; j < 4; ++j) {
                BFrag16 bf;
                const int col = Noff + j * 16 + ml;
                const uint4* p = (const uint4*)&bs[col * BK + ks + bbase];
                bf.q[0] = p[0];
                bf.q[1] = p[2];
#pragma unroll
                for (int i = 0; i < 4; ++i)
                    c[i][j] = __builtin_amdgcn_wmma_f32_16x16x32_bf16(
                        false, af[i].v, false, bf.v, (short)0, c[i][j], false, false);
            }
        }
        __syncthreads();
    }

    // epilogue: +b1, exact GELU, bf16 store
#pragma unroll
    for (int i = 0; i < 4; ++i)
#pragma unroll
        for (int j = 0; j < 4; ++j) {
            const int n = n0 + Noff + j * 16 + ml;
            const float bias = b1[n];
#pragma unroll
            for (int v = 0; v < 8; ++v) {
                const int m = m0 + Moff + i * 16 + (half ? v + 8 : v);
                union { __bf16 b; unsigned short u; } cv;
                cv.b = (__bf16)gelu_exact(c[i][j][v] + bias);
                H[(size_t)m * DIN + n] = cv.u;
            }
        }
}

// ---------------------------------------------------------------------------
// Kernel 2: logits = H @ W2 + b2 (WMMA), per-token top-8 softmax routing,
// load-balance accumulators. 128 tokens per block (8 waves x 16 tokens).
// ---------------------------------------------------------------------------
__global__ __launch_bounds__(256) void router_kernel(const unsigned short* __restrict__ H,
                                                     const unsigned short* __restrict__ w2t,
                                                     const float* __restrict__ b2,
                                                     float* __restrict__ rw_out,
                                                     float* __restrict__ idx_out,
                                                     float* __restrict__ probs_acc,
                                                     float* __restrict__ cnt_acc) {
    __shared__ float lg[8 * 16 * 64];   // per-wave 16x64 logits, 32 KB
    __shared__ float pacc[NEXP];
    __shared__ unsigned cacc[NEXP];

    const int tid  = threadIdx.x;
    const int lane = tid & 31;
    const int w    = tid >> 5;
    const int ml    = lane & 15;
    const int half  = lane >> 4;
    const int abase = half * 8;
    const int bbase = half * 8;

    if (tid < NEXP) { pacc[tid] = 0.0f; cacc[tid] = 0u; }

    const int tok0 = blockIdx.x * 128 + w * 16;
    const unsigned short* hrow = H + (size_t)(tok0 + ml) * DIN;

    v8f c[4] = {};
    for (int k0 = 0; k0 < DIN; k0 += 32) {
        BFrag16 a;
        a.q[0] = *(const uint4*)(hrow + k0 + abase);
        a.q[1] = *(const uint4*)(hrow + k0 + 16 + abase);
#pragma unroll
        for (int j = 0; j < 4; ++j) {
            BFrag16 b;
            const uint4* p = (const uint4*)(w2t + (size_t)(j * 16 + ml) * DIN + k0 + bbase);
            b.q[0] = p[0];
            b.q[1] = p[2];
            c[j] = __builtin_amdgcn_wmma_f32_16x16x32_bf16(
                false, a.v, false, b.v, (short)0, c[j], false, false);
        }
    }

#pragma unroll
    for (int j = 0; j < 4; ++j) {
        const int e = j * 16 + ml;
        const float bias = b2[e];
#pragma unroll
        for (int v = 0; v < 8; ++v) {
            const int m = half ? v + 8 : v;
            lg[w * 1024 + m * 64 + e] = c[j][v] + bias;
        }
    }
    __syncthreads();

    if (tid < 128) {
        const int token = blockIdx.x * 128 + tid;
        const float* L = &lg[(tid >> 4) * 1024 + (tid & 15) * 64];

        // full softmax accumulation (for avg_prob)
        float mx = -3.4e38f;
        for (int e = 0; e < NEXP; ++e) mx = fmaxf(mx, L[e]);
        float s = 0.0f;
        for (int e = 0; e < NEXP; ++e) s += __expf(L[e] - mx);
        const float inv_s = 1.0f / s;
        for (int e = 0; e < NEXP; ++e)
            atomicAdd(&pacc[e], __expf(L[e] - mx) * inv_s);

        // top-8 selection (ties -> lowest index, matches lax.top_k)
        unsigned long long used = 0ull;
        float tv[KTOP];
        int   ti[KTOP];
#pragma unroll
        for (int k = 0; k < KTOP; ++k) {
            float best = -3.4e38f;
            int   bi   = 0;
            for (int e = 0; e < NEXP; ++e) {
                if (!((used >> e) & 1ull) && L[e] > best) { best = L[e]; bi = e; }
            }
            used |= 1ull << bi;
            tv[k] = best;
            ti[k] = bi;
        }
        float s8 = 0.0f;
#pragma unroll
        for (int k = 0; k < KTOP; ++k) s8 += __expf(tv[k] - tv[0]);
        const float inv_s8 = 1.0f / s8;

        float* rw = rw_out + (size_t)token * NEXP;
        for (int e = 0; e < NEXP; ++e) rw[e] = 0.0f;
#pragma unroll
        for (int k = 0; k < KTOP; ++k) {
            rw[ti[k]] = __expf(tv[k] - tv[0]) * inv_s8;
            idx_out[(size_t)token * KTOP + k] = (float)ti[k];
            atomicAdd(&cacc[ti[k]], 1u);
        }
    }
    __syncthreads();

    if (tid < NEXP) {
        atomicAdd(&probs_acc[tid], pacc[tid]);
        atomicAdd(&cnt_acc[tid], (float)cacc[tid]);
    }
}

// ---------------------------------------------------------------------------
// Kernel 3: loss = E * sum_e (cnt[e]/N) * (probsum[e]/N)
// ---------------------------------------------------------------------------
__global__ __launch_bounds__(64) void loss_kernel(const float* __restrict__ probs_acc,
                                                  const float* __restrict__ cnt_acc,
                                                  float* __restrict__ out_loss) {
    __shared__ float red[NEXP];
    const int t = threadIdx.x;
    red[t] = probs_acc[t] * cnt_acc[t];
    __syncthreads();
    for (int sft = 32; sft > 0; sft >>= 1) {
        if (t < sft) red[t] += red[t + sft];
        __syncthreads();
    }
    if (t == 0)
        *out_loss = (float)NEXP * red[0] / ((float)NTOK * (float)NTOK);
}

// ---------------------------------------------------------------------------
extern "C" void kernel_launch(void* const* d_in, const int* in_sizes, int n_in,
                              void* d_out, int out_size, void* d_ws, size_t ws_size,
                              hipStream_t stream) {
    const float* x  = (const float*)d_in[0];   // [N, D]
    const float* W1 = (const float*)d_in[1];   // [D, D]
    const float* b1 = (const float*)d_in[2];   // [D]
    const float* W2 = (const float*)d_in[3];   // [D, E]
    const float* b2 = (const float*)d_in[4];   // [E]

    float* rw_out   = (float*)d_out;                         // [N, E]
    float* idx_out  = rw_out + (size_t)NTOK * NEXP;          // [N, K] (as float)
    float* loss_out = idx_out + (size_t)NTOK * KTOP;         // [1]

    char* ws = (char*)d_ws;
    size_t off = 0;
    unsigned short* xbuf = (unsigned short*)(ws + off); off += (size_t)NTOK * DIN * 2;  // 64 MB
    unsigned short* w1t  = (unsigned short*)(ws + off); off += (size_t)DIN  * DIN * 2;  //  8 MB
    unsigned short* Hbuf = (unsigned short*)(ws + off); off += (size_t)NTOK * DIN * 2;  // 64 MB
    unsigned short* w2t  = (unsigned short*)(ws + off); off += (size_t)NEXP * DIN * 2;  // 256 KB
    float* probs_acc = (float*)(ws + off);
    float* cnt_acc   = probs_acc + NEXP;

    hipMemsetAsync(probs_acc, 0, 2 * NEXP * sizeof(float), stream);

    cvt_x_kernel  <<<(NTOK * (DIN / 8)) / 256, 256, 0, stream>>>(x, xbuf);
    cvt_w1t_kernel<<<dim3(DIN / 32, DIN / 32), 256, 0, stream>>>(W1, w1t);
    cvt_w2t_kernel<<<(NEXP * DIN) / 256, 256, 0, stream>>>(W2, w2t);

    gemm1_gelu<<<dim3(NTOK / BM, DIN / BN), 256, 0, stream>>>(xbuf, w1t, b1, Hbuf);
    router_kernel<<<NTOK / 128, 256, 0, stream>>>(Hbuf, w2t, b2, rw_out, idx_out,
                                                  probs_acc, cnt_acc);
    loss_kernel<<<1, 64, 0, stream>>>(probs_acc, cnt_acc, loss_out);
}
